// Transformer_37778532335769
// MI455X (gfx1250) — matrix operand order repaired and checked by reading, not compiled, not run on hardware
//
#include <hip/hip_runtime.h>

// ---------------------------------------------------------------- constants
static constexpr int DIMC  = 512;
static constexpr int HC    = 8;
static constexpr int DHC   = 64;
static constexpr int INNERC= 2048;
static constexpr int SEQC  = 2048;
static constexpr int NLAY  = 6;

static constexpr int BM = 128;       // block tile M (all GEMMs here have M=2048)
static constexpr int BN = 64;        // block tile N
static constexpr int BK = 32;        // K step (one bf16 WMMA K)
static constexpr int LDP = BK + 8;   // padded LDS row (ushorts); 80B => 16B-aligned rows

typedef __attribute__((ext_vector_type(16))) __bf16 bf16x16;
typedef __attribute__((ext_vector_type(8)))  float  f32x8;
typedef unsigned short u16;
typedef int v4i __attribute__((vector_size(16)));

#define GLOBAL_AS __attribute__((address_space(1)))
#define LDS_AS    __attribute__((address_space(3)))

union Frag16 { bf16x16 v; unsigned int u[8]; };

// async global->LDS path (gfx1250), guarded so compile can't break
#if defined(__has_builtin)
#if __has_builtin(__builtin_amdgcn_global_load_async_to_lds_b128) && \
    __has_builtin(__builtin_amdgcn_s_wait_asynccnt)
#define USE_ASYNC_LDS 1
#endif
#endif
#ifndef USE_ASYNC_LDS
#define USE_ASYNC_LDS 0
#endif

__device__ __forceinline__ u16 f2bf(float f) {
  union { float f; unsigned int u; } x; x.f = f;
  unsigned int r = 0x7FFFu + ((x.u >> 16) & 1u);   // round-to-nearest-even
  return (u16)((x.u + r) >> 16);
}
__device__ __forceinline__ float bf2f(u16 s) {
  union { unsigned int u; float f; } x; x.u = ((unsigned int)s) << 16; return x.f;
}

// 16-byte tile chunk copy: async DMA to LDS if available, else vector copy
__device__ __forceinline__ void copy16(u16* l, const u16* g) {
#if USE_ASYNC_LDS
  __builtin_amdgcn_global_load_async_to_lds_b128(
      (GLOBAL_AS v4i*)(v4i*)(void*)g,
      (LDS_AS v4i*)(v4i*)(void*)l, 0, 0);
#else
  *(uint4*)l = *(const uint4*)g;
#endif
}
__device__ __forceinline__ void wait_async() {
#if USE_ASYNC_LDS
  __builtin_amdgcn_s_wait_asynccnt(0);
#endif
}

// ---------------------------------------------------------------- GEMM (WMMA, bf16 in, f32 acc)
// C[M,N] = alpha * A[M,K] * op(B) (+bias[n]) (+res[m,n]) (relu?), batched over z.
// op(B) = B[K,N] (transB=0) or B[N,K]^T (transB=1). A,B bf16; C written as f32
// (Cf) and/or bf16 (Cb). Double-buffered LDS: async DMA of tile k+1 overlaps
// the 8 WMMAs of tile k (one barrier + one asynccnt wait per iteration).
__launch_bounds__(128)
__global__ void gemm_bf16(int K,
                          const u16* __restrict__ A, int lda, long long sA,
                          const u16* __restrict__ B, int ldb, long long sB, int transB,
                          float* __restrict__ Cf, u16* __restrict__ Cb,
                          int ldc, long long sC,
                          float alpha,
                          const float* __restrict__ bias,
                          const float* __restrict__ res, int ldr, int relu) {
  __shared__ __align__(16) u16 As[2][BM * LDP];   // As[m][k], 2 buffers
  __shared__ __align__(16) u16 Bs[2][BN * LDP];   // Bs[n][k], 2 buffers

  const int z = blockIdx.z;
  A += (long long)z * sA;
  B += (long long)z * sB;

  const int n0 = blockIdx.x * BN;
  const int m0 = blockIdx.y * BM;
  const int tid   = threadIdx.x;
  const int lane  = tid & 31;
  const int wave  = tid >> 5;          // wave owns rows [wave*32, wave*32+32)
  const int lhalf = lane >> 4;
  const int l15   = lane & 15;

  auto stageA = [&](int buf, int k0) {
#pragma unroll
    for (int h = 0; h < 4; ++h) {       // 512 chunks of 16B / 128 threads
      int c = h * 128 + tid;
      int r = c >> 2, cc = c & 3;
      copy16(&As[buf][r * LDP + cc * 8],
             &A[(long long)(m0 + r) * lda + k0 + cc * 8]);
    }
  };
  auto stageB = [&](int buf, int k0) {
    if (transB) {
      // B[N,K]: rows already k-contiguous -> async 16B chunks
#pragma unroll
      for (int h = 0; h < 2; ++h) {
        int c = h * 128 + tid;
        int r = c >> 2, cc = c & 3;
        copy16(&Bs[buf][r * LDP + cc * 8],
               &B[(long long)(n0 + r) * ldb + k0 + cc * 8]);
      }
    } else {
      // B[K,N]: transpose during staging (coalesced 8B loads, b16 scatters)
#pragma unroll
      for (int h = 0; h < 4; ++h) {
        int c  = h * 128 + tid;          // 0..511
        int kr = c >> 4, n4 = (c & 15) * 4;
        unsigned long long w =
            *(const unsigned long long*)&B[(long long)(k0 + kr) * ldb + n0 + n4];
        Bs[buf][(n4 + 0) * LDP + kr] = (u16)(w);
        Bs[buf][(n4 + 1) * LDP + kr] = (u16)(w >> 16);
        Bs[buf][(n4 + 2) * LDP + kr] = (u16)(w >> 32);
        Bs[buf][(n4 + 3) * LDP + kr] = (u16)(w >> 48);
      }
      if (k0 + BK < K)  // L2 prefetch for the tile after next (global_prefetch_b8)
        __builtin_prefetch((const void*)&B[(long long)(k0 + BK + (tid & 31)) * ldb + n0], 0, 1);
    }
  };

  f32x8 acc[2][4] = {};

  stageA(0, 0);
  stageB(0, 0);
  int cur = 0;
  for (int k0 = 0; k0 < K; k0 += BK) {
    wait_async();
    __syncthreads();                     // buffer `cur` complete & visible
    if (k0 + BK < K) {                   // overlap next tile's DMA with compute
      stageA(cur ^ 1, k0 + BK);
      stageB(cur ^ 1, k0 + BK);
    }

    // A fragments (ISA 16-bit A 16x32 layout: lane half picks K interleave)
    Frag16 a0, a1;
#pragma unroll
    for (int j = 0; j < 8; ++j) {
      int kb = (j < 4 ? 2 * j : 16 + 2 * (j - 4)) + lhalf * 8;
      a0.u[j] = *(const unsigned int*)&As[cur][(wave * 32 +      l15) * LDP + kb];
      a1.u[j] = *(const unsigned int*)&As[cur][(wave * 32 + 16 + l15) * LDP + kb];
    }
#pragma unroll
    for (int t = 0; t < 4; ++t) {
      // B fragment: lane = column, lane half picks K interleave (same as A)
      Frag16 b;
      const int col  = t * 16 + l15;
      const int koff = lhalf * 8;
#pragma unroll
      for (int j = 0; j < 8; ++j)
        b.u[j] = *(const unsigned int*)&Bs[cur][col * LDP + koff + (j < 4 ? 2 * j : 16 + 2 * (j - 4))];
      acc[0][t] = __builtin_amdgcn_wmma_f32_16x16x32_bf16(
          false, a0.v, false, b.v, (short)0, acc[0][t], false, false);
      acc[1][t] = __builtin_amdgcn_wmma_f32_16x16x32_bf16(
          false, a1.v, false, b.v, (short)0, acc[1][t], false, false);
    }
    cur ^= 1;
  }

  // epilogue (D layout: lane 0-15 -> M=r, lane 16-31 -> M=r+8; N = lane&15)
  float* Cfz = Cf ? Cf + (long long)z * sC : nullptr;
  u16*   Cbz = Cb ? Cb + (long long)z * sC : nullptr;
#pragma unroll
  for (int s = 0; s < 2; ++s) {
#pragma unroll
    for (int t = 0; t < 4; ++t) {
#pragma unroll
      for (int r = 0; r < 8; ++r) {
        int m = m0 + wave * 32 + s * 16 + r + lhalf * 8;
        int n = n0 + t * 16 + l15;
        float v = alpha * acc[s][t][r];
        if (bias) v += bias[n];
        if (res)  v += res[(long long)m * ldr + n];
        if (relu) v = fmaxf(v, 0.0f);
        if (Cfz) Cfz[(long long)m * ldc + n] = v;
        if (Cbz) Cbz[(long long)m * ldc + n] = f2bf(v);
      }
    }
  }
}

// ---------------------------------------------------------------- helpers
__global__ void embed_pos(const int* __restrict__ ids, const float* __restrict__ emb,
                          float* __restrict__ out, u16* __restrict__ outb) {
  int n = blockIdx.x, d = threadIdx.x;                 // blockDim = 512
  float ang = (float)n * __expf(-(float)d * (9.210340371976184f / 256.0f));
  float pe  = (d & 1) ? __cosf(ang) : __sinf(ang);
  float v   = emb[(size_t)ids[n] * DIMC + d] + pe;
  out [(size_t)n * DIMC + d] = v;
  outb[(size_t)n * DIMC + d] = f2bf(v);
}

__global__ void f2bf_kernel(const float* __restrict__ in, u16* __restrict__ out, int n) {
  int idx = blockIdx.x * 256 + threadIdx.x;
  if (idx < n) out[idx] = f2bf(in[idx]);
}

__global__ void triu_mask_bf(const float* __restrict__ in, u16* __restrict__ out, int total) {
  int idx = blockIdx.x * 256 + threadIdx.x;
  if (idx >= total) return;
  int rc = idx % (DIMC * DHC);
  int r = rc / DHC, c = rc % DHC;
  out[idx] = f2bf((c >= r) ? in[idx] : 0.0f);
}

// in-place softmax over bf16 rows of length SEQC (256 thr, 8 elems/thr)
__global__ void softmax_rows_bf(u16* __restrict__ x) {
  u16* row = x + (size_t)blockIdx.x * SEQC;
  __shared__ float red[256];
  const int tid = threadIdx.x;
  float v[8];
#pragma unroll
  for (int i = 0; i < 8; ++i) v[i] = bf2f(row[tid + 256 * i]);
  float m = v[0];
#pragma unroll
  for (int i = 1; i < 8; ++i) m = fmaxf(m, v[i]);
  red[tid] = m; __syncthreads();
  for (int o = 128; o > 0; o >>= 1) { if (tid < o) red[tid] = fmaxf(red[tid], red[tid + o]); __syncthreads(); }
  m = red[0]; __syncthreads();
  float s = 0.0f;
#pragma unroll
  for (int i = 0; i < 8; ++i) { v[i] = __expf(v[i] - m); s += v[i]; }
  red[tid] = s; __syncthreads();
  for (int o = 128; o > 0; o >>= 1) { if (tid < o) red[tid] += red[tid + o]; __syncthreads(); }
  float inv = 1.0f / red[0];
#pragma unroll
  for (int i = 0; i < 8; ++i) row[tid + 256 * i] = f2bf(v[i] * inv);
}

__global__ void layernorm_k(const float* __restrict__ in, float* __restrict__ out,
                            u16* __restrict__ outb,
                            const float* __restrict__ g, const float* __restrict__ b) {
  const int row = blockIdx.x, tid = threadIdx.x;       // blockDim = 128
  const float* x = in + (size_t)row * DIMC;
  __shared__ float red[128];
  float s = 0.0f;
  for (int i = tid; i < DIMC; i += 128) s += x[i];
  red[tid] = s; __syncthreads();
  for (int o = 64; o > 0; o >>= 1) { if (tid < o) red[tid] += red[tid + o]; __syncthreads(); }
  float mu = red[0] * (1.0f / DIMC); __syncthreads();
  float v = 0.0f;
  for (int i = tid; i < DIMC; i += 128) { float d = x[i] - mu; v += d * d; }
  red[tid] = v; __syncthreads();
  for (int o = 64; o > 0; o >>= 1) { if (tid < o) red[tid] += red[tid + o]; __syncthreads(); }
  float inv = rsqrtf(red[0] * (1.0f / DIMC) + 1e-5f);
  for (int i = tid; i < DIMC; i += 128) {
    float r = g[i] * (x[i] - mu) * inv + b[i];
    out [(size_t)row * DIMC + i] = r;
    outb[(size_t)row * DIMC + i] = f2bf(r);
  }
}

// ---------------------------------------------------------------- launch
extern "C" void kernel_launch(void* const* d_in, const int* in_sizes, int n_in,
                              void* d_out, int out_size, void* d_ws, size_t ws_size,
                              hipStream_t stream) {
  (void)in_sizes; (void)n_in; (void)out_size; (void)ws_size;

  const int*   x     = (const int*)d_in[0];
  const int*   y     = (const int*)d_in[1];
  const float* emb   = (const float*)d_in[2];
  const float* eWq   = (const float*)d_in[3];
  const float* eWk   = (const float*)d_in[4];
  const float* eWv   = (const float*)d_in[5];
  const float* eWo   = (const float*)d_in[6];
  const float* eln1g = (const float*)d_in[7];
  const float* eln1b = (const float*)d_in[8];
  const float* eln2g = (const float*)d_in[9];
  const float* eln2b = (const float*)d_in[10];
  const float* ew1   = (const float*)d_in[11];
  const float* eb1   = (const float*)d_in[12];
  const float* ew2   = (const float*)d_in[13];
  const float* eb2   = (const float*)d_in[14];
  const float* dWq1  = (const float*)d_in[15];
  const float* dWk1  = (const float*)d_in[16];
  const float* dWv1  = (const float*)d_in[17];
  const float* dWo1  = (const float*)d_in[18];
  const float* dWq2  = (const float*)d_in[19];
  const float* dWk2  = (const float*)d_in[20];
  const float* dWv2  = (const float*)d_in[21];
  const float* dWo2  = (const float*)d_in[22];
  const float* dln1g = (const float*)d_in[23];
  const float* dln1b = (const float*)d_in[24];
  const float* dln2g = (const float*)d_in[25];
  const float* dln2b = (const float*)d_in[26];
  const float* dw1   = (const float*)d_in[27];
  const float* db1   = (const float*)d_in[28];
  const float* dw2   = (const float*)d_in[29];
  const float* db2   = (const float*)d_in[30];

  // ---------- workspace carve-out (256B-aligned chunks) ----------
  char* base = (char*)d_ws;
  size_t off = 0;
  auto allocF = [&](size_t n) { float* p = (float*)(base + off); off = (off + n * 4 + 255) & ~(size_t)255; return p; };
  auto allocB = [&](size_t n) { u16*   p = (u16*)  (base + off); off = (off + n * 2 + 255) & ~(size_t)255; return p; };

  // f32 state
  float* e_buf = allocF((size_t)SEQC * DIMC);
  float* d_buf = allocF((size_t)SEQC * DIMC);
  float* t0f   = allocF((size_t)SEQC * DIMC);      // pre-LN feed
  float* t1    = allocF((size_t)SEQC * DIMC);      // out1 / out2
  float* t2    = allocF((size_t)SEQC * DIMC);      // norm1
  // bf16 activations
  u16* e_bf  = allocB((size_t)SEQC * DIMC);
  u16* d_bf  = allocB((size_t)SEQC * DIMC);
  u16* t1_bf = allocB((size_t)SEQC * DIMC);
  u16* t2_bf = allocB((size_t)SEQC * DIMC);
  u16* t0_bf = allocB((size_t)SEQC * DIMC);        // att concat out
  u16* t3_bf = allocB((size_t)SEQC * INNERC);      // ffn hidden
  u16* qh    = allocB((size_t)SEQC * DIMC);        // [H][SEQ][DH]
  u16* kh    = allocB((size_t)SEQC * DIMC);
  u16* vh    = allocB((size_t)SEQC * DIMC);
  u16* att   = allocB((size_t)HC * SEQC * SEQC);   // 67 MB, bf16 scores/attention
  // bf16 weights
  const size_t WP = (size_t)HC * DIMC * DHC, WO = (size_t)DIMC * DIMC, WF = (size_t)DIMC * INNERC;
  u16* ewq_b = allocB(WP); u16* ewk_b = allocB(WP); u16* ewv_b = allocB(WP); u16* ewo_b = allocB(WO);
  u16* ew1_b = allocB(WF); u16* ew2_b = allocB(WF);
  u16* wq1m  = allocB(WP); u16* wk1m  = allocB(WP); u16* wv1m  = allocB(WP); u16* dwo1_b = allocB(WO);
  u16* dwq2_b = allocB(WP); u16* dwk2_b = allocB(WP); u16* dwv2_b = allocB(WP); u16* dwo2_b = allocB(WO);
  u16* dw1_b = allocB(WF); u16* dw2_b = allocB(WF);

  auto gemm = [&](int M, int N, int K,
                  const u16* A, int lda, long long sA,
                  const u16* B, int ldb, long long sB, int tB,
                  float* Cf, u16* Cb, int ldc, long long sC, int nb, float alpha,
                  const float* bias, const float* res, int ldr, int relu) {
    gemm_bf16<<<dim3(N / BN, M / BM, nb), 128, 0, stream>>>(
        K, A, lda, sA, B, ldb, sB, tB, Cf, Cb, ldc, sC, alpha, bias, res, ldr, relu);
  };
  auto conv = [&](const float* in, u16* out, size_t n) {
    f2bf_kernel<<<((int)n + 255) / 256, 256, 0, stream>>>(in, out, (int)n);
  };

  // MHA in bf16: QKV proj -> scores(NT, alpha=1/8) -> softmax -> att*V (concat
  // layout) -> out = concat*Wo + resid (f32, optional bf16 copy)
  auto mha = [&](const u16* qin, const u16* kvin,
                 const u16* Wq, const u16* Wk, const u16* Wv, const u16* Wo,
                 const float* resid, float* outF, u16* outB) {
    const long long sw = (long long)DIMC * DHC, sh = (long long)SEQC * DHC;
    const long long ss = (long long)SEQC * SEQC;
    gemm(SEQC, DHC, DIMC, qin,  DIMC, 0, Wq, DHC, sw, 0, nullptr, qh, DHC, sh, HC, 1.0f, nullptr, nullptr, 0, 0);
    gemm(SEQC, DHC, DIMC, kvin, DIMC, 0, Wk, DHC, sw, 0, nullptr, kh, DHC, sh, HC, 1.0f, nullptr, nullptr, 0, 0);
    gemm(SEQC, DHC, DIMC, kvin, DIMC, 0, Wv, DHC, sw, 0, nullptr, vh, DHC, sh, HC, 1.0f, nullptr, nullptr, 0, 0);
    gemm(SEQC, SEQC, DHC, qh, DHC, sh, kh, DHC, sh, 1,
         nullptr, att, SEQC, ss, HC, 0.125f, nullptr, nullptr, 0, 0);
    softmax_rows_bf<<<HC * SEQC, 256, 0, stream>>>(att);
    gemm(SEQC, DHC, SEQC, att, SEQC, ss, vh, DHC, sh, 0,
         nullptr, t0_bf, DIMC, (long long)DHC, HC, 1.0f, nullptr, nullptr, 0, 0);
    gemm(SEQC, DIMC, DIMC, t0_bf, DIMC, 0, Wo, DIMC, 0, 0,
         outF, outB, DIMC, 0, 1, 1.0f, nullptr, resid, DIMC, 0);
  };

  auto ffn_ln = [&](const u16* in_bf, const float* resid,
                    const u16* w1, const float* b1, const u16* w2, const float* b2,
                    const float* g, const float* bb, float* outF, u16* outB) {
    gemm(SEQC, INNERC, DIMC, in_bf, DIMC, 0, w1, INNERC, 0, 0,
         nullptr, t3_bf, INNERC, 0, 1, 1.0f, b1, nullptr, 0, 1);
    gemm(SEQC, DIMC, INNERC, t3_bf, INNERC, 0, w2, DIMC, 0, 0,
         t0f, nullptr, DIMC, 0, 1, 1.0f, b2, resid, DIMC, 0);
    layernorm_k<<<SEQC, 128, 0, stream>>>(t0f, outF, outB, g, bb);
  };

  // ---- one-time weight conversion to bf16 (+ triu quirk for decoder self-attn) ----
  conv(eWq, ewq_b, WP);  conv(eWk, ewk_b, WP);  conv(eWv, ewv_b, WP);  conv(eWo, ewo_b, WO);
  conv(ew1, ew1_b, WF);  conv(ew2, ew2_b, WF);
  conv(dWo1, dwo1_b, WO);
  conv(dWq2, dwq2_b, WP); conv(dWk2, dwk2_b, WP); conv(dWv2, dwv2_b, WP); conv(dWo2, dwo2_b, WO);
  conv(dw1, dw1_b, WF);  conv(dw2, dw2_b, WF);
  triu_mask_bf<<<((int)WP + 255) / 256, 256, 0, stream>>>(dWq1, wq1m, (int)WP);
  triu_mask_bf<<<((int)WP + 255) / 256, 256, 0, stream>>>(dWk1, wk1m, (int)WP);
  triu_mask_bf<<<((int)WP + 255) / 256, 256, 0, stream>>>(dWv1, wv1m, (int)WP);

  // ---- embeddings + positional encoding ----
  embed_pos<<<SEQC, DIMC, 0, stream>>>(x, emb, e_buf, e_bf);
  embed_pos<<<SEQC, DIMC, 0, stream>>>(y, emb, d_buf, d_bf);

  for (int l = 0; l < NLAY; ++l) {
    // ---- encoder layer ----
    mha(e_bf, e_bf, ewq_b, ewk_b, ewv_b, ewo_b, e_buf, t1, nullptr);
    layernorm_k<<<SEQC, 128, 0, stream>>>(t1, t2, t2_bf, eln1g, eln1b);
    ffn_ln(t2_bf, t1, ew1_b, eb1, ew2_b, eb2, eln2g, eln2b, e_buf, e_bf);

    // ---- decoder layer (uses updated encoder state) ----
    mha(d_bf, d_bf, wq1m, wk1m, wv1m, dwo1_b, d_buf, t1, nullptr);
    layernorm_k<<<SEQC, 128, 0, stream>>>(t1, t2, t2_bf, dln1g, dln1b);
    mha(t2_bf, e_bf, dwq2_b, dwk2_b, dwv2_b, dwo2_b, t2, t1, t1_bf);
    ffn_ln(t1_bf, t1, dw1_b, db1, dw2_b, db2, dln2g, dln2b, d_buf, d_bf);
  }

  (void)hipMemcpyAsync(d_out, d_buf, sizeof(float) * (size_t)SEQC * DIMC,
                       hipMemcpyDeviceToDevice, stream);
}
